// ChannelLinearAttention_57810259804533
// MI455X (gfx1250) — matrix operand order repaired and checked by reading, not compiled, not run on hardware
//
#include <hip/hip_runtime.h>
#include <hip/hip_bf16.h>

typedef __bf16 bf16;
typedef __attribute__((ext_vector_type(16))) bf16  v16bf;
typedef __attribute__((ext_vector_type(8)))  bf16  v8bf;
typedef __attribute__((ext_vector_type(8)))  float v8f;
typedef __attribute__((ext_vector_type(4)))  float v4f;

constexpr int Bb = 16;    // batch
constexpr int Ll = 4096;  // sequence
constexpr int HE = 512;   // H*E channels
constexpr float EPSV = 1e-6f;

static __device__ __forceinline__ v16bf cat8(v8bf lo, v8bf hi) {
  return __builtin_shufflevector(lo, hi, 0,1,2,3,4,5,6,7,8,9,10,11,12,13,14,15);
}

static __device__ __forceinline__ void wait_asynccnt0() {
#if __has_builtin(__builtin_amdgcn_s_wait_asynccnt)
  __builtin_amdgcn_s_wait_asynccnt(0);
#else
  asm volatile("s_wait_asynccnt 0x0" ::: "memory");
#endif
}

// ---------------------------------------------------------------------------
// Kernel 1a: partial column sums / sums-of-squares over L-chunks.
// grid (B, 8), block 128; thread owns 4 consecutive columns (float4 coalesced).
// ---------------------------------------------------------------------------
__global__ __launch_bounds__(128)
void stats_part_k(const float* __restrict__ V, float* __restrict__ psum,
                  float* __restrict__ psumsq) {
  const int b = blockIdx.x, ch = blockIdx.y;
  const int n4 = threadIdx.x * 4;
  const float* base = V + ((size_t)b * Ll + (size_t)ch * (Ll / 8)) * HE + n4;
  v4f s  = {0.f, 0.f, 0.f, 0.f};
  v4f ss = {0.f, 0.f, 0.f, 0.f};
#pragma unroll 4
  for (int l = 0; l < Ll / 8; ++l) {
    v4f v = *(const v4f*)(base + (size_t)l * HE);
    s += v;
    ss += v * v;
  }
  const size_t o = ((size_t)(b * 8 + ch)) * HE + n4;
  *(v4f*)&psum[o]   = s;
  *(v4f*)&psumsq[o] = ss;
}

// ---------------------------------------------------------------------------
// Kernel 1b: finalize stats -> norm_inv, vsum, c  (c[n] = (vsum/norm + eps)/norm)
// ---------------------------------------------------------------------------
__global__ __launch_bounds__(128)
void stats_final_k(const float* __restrict__ psum, const float* __restrict__ psumsq,
                   float* __restrict__ norm_inv, float* __restrict__ vsum,
                   float* __restrict__ cvec) {
  const int b = blockIdx.x;
  const int n4 = threadIdx.x * 4;
  v4f s  = {0.f, 0.f, 0.f, 0.f};
  v4f ss = {0.f, 0.f, 0.f, 0.f};
#pragma unroll
  for (int ch = 0; ch < 8; ++ch) {
    const size_t o = ((size_t)(b * 8 + ch)) * HE + n4;
    s  += *(const v4f*)&psum[o];
    ss += *(const v4f*)&psumsq[o];
  }
#pragma unroll
  for (int i = 0; i < 4; ++i) {
    const int n = n4 + i;
    const float ninv = 1.0f / sqrtf(ss[i]);
    norm_inv[b * HE + n] = ninv;
    vsum[b * HE + n]     = s[i];
    cvec[b * HE + n]     = ninv * (s[i] * ninv + EPSV);
  }
}

// ---------------------------------------------------------------------------
// Kernel 2: Gram matrix  KQ[b,n,m] = norm_inv[n]*norm_inv[m] * sum_l V[l,n]V[l,m]
// Symmetric: only (bi<=bj) 64x64 blocks computed; mirror written.
// LDS tiles stored TRANSPOSED (channel-major) so WMMA operand fragments are
// contiguous ds_read_b128s. block = 128 threads = 4 waves, each wave does a
// 32x32 quadrant (4 accumulators) of the 64x64 block.
// ---------------------------------------------------------------------------
__global__ __launch_bounds__(128)
void gram_k(const float* __restrict__ V, const float* __restrict__ norm_inv,
            bf16* __restrict__ KQ) {
  constexpr int LP = 48;  // padded LDS row stride (bf16 elems), 96B (16B-aligned)
  __shared__ bf16 ldsA[64][LP];
  __shared__ bf16 ldsB[64][LP];

  const int b = blockIdx.x;
  // map linear pair index -> upper-triangular (bi, bj) over 8x8 blocks of 64
  int bi = 0, t = (int)blockIdx.y;
  while (t >= 8 - bi) { t -= 8 - bi; ++bi; }
  const int bj = bi + t;
  const int n0 = bi * 64, m0 = bj * 64;

  const int tid  = threadIdx.x;
  const int wave = tid >> 5, lane = tid & 31;
  const int lr = lane & 15, hh = lane >> 4;
  const int wn = (wave >> 1) * 32;  // wave's n offset within block
  const int wm = (wave & 1) * 32;   // wave's m offset within block

  v8f acc[2][2];
#pragma unroll
  for (int i = 0; i < 2; ++i)
#pragma unroll
    for (int j = 0; j < 2; ++j) acc[i][j] = {0.f,0.f,0.f,0.f,0.f,0.f,0.f,0.f};

  // cooperative-load assignment: 64 threads per tile, one column each
  const int col   = tid & 63;
  const int which = tid >> 6;  // 0 -> A tile (n cols), 1 -> B tile (m cols)
  const int gcol  = (which ? m0 : n0) + col;
  bf16* dst = which ? &ldsB[col][0] : &ldsA[col][0];

  for (int ls = 0; ls < Ll / 32; ++ls) {
    const int lbase = ls * 32;
    const float* src = V + ((size_t)b * Ll + lbase) * HE + gcol;
#pragma unroll 8
    for (int l = 0; l < 32; ++l) dst[l] = (bf16)src[(size_t)l * HE];
    __syncthreads();

    v16bf aop[2], bop[2];
#pragma unroll
    for (int s = 0; s < 2; ++s) {
      const int row = wn + s * 16 + lr;          // A: M index (n)
      v8bf alo = *(const v8bf*)&ldsA[row][hh * 8];
      v8bf ahi = *(const v8bf*)&ldsA[row][hh * 8 + 16];
      aop[s] = cat8(alo, ahi);
      const int cm = wm + s * 16 + lr;           // B: N index (m)
      v8bf blo = *(const v8bf*)&ldsB[cm][hh * 8];
      v8bf bhi = *(const v8bf*)&ldsB[cm][hh * 8 + 16];
      bop[s] = cat8(blo, bhi);
    }
#pragma unroll
    for (int sn = 0; sn < 2; ++sn)
#pragma unroll
      for (int sm = 0; sm < 2; ++sm)
        acc[sn][sm] = __builtin_amdgcn_wmma_f32_16x16x32_bf16(
            false, aop[sn], false, bop[sm], (short)0, acc[sn][sm], false, false);
    __syncthreads();
  }

  // epilogue: scale by norm_inv[n]*norm_inv[m], store bf16 (+ mirror)
#pragma unroll
  for (int sn = 0; sn < 2; ++sn)
#pragma unroll
    for (int sm = 0; sm < 2; ++sm) {
      const int m = m0 + wm + sm * 16 + lr;
      const float nim = norm_inv[b * HE + m];
#pragma unroll
      for (int r = 0; r < 8; ++r) {
        const int n = n0 + wn + sn * 16 + r + 8 * hh;
        const float val = acc[sn][sm][r] * norm_inv[b * HE + n] * nim;
        KQ[((size_t)b * HE + n) * HE + m] = (bf16)val;
        if (bi != bj) KQ[((size_t)b * HE + m) * HE + n] = (bf16)val;
      }
    }
}

// ---------------------------------------------------------------------------
// Kernel 3: tailor[b,l] = 1 / (512 + dot(V[b,l,:], c[b,:]))
// wave per row; lane covers 16 consecutive channels; shfl-xor reduction.
// ---------------------------------------------------------------------------
__global__ __launch_bounds__(256)
void tailor_k(const float* __restrict__ V, const float* __restrict__ cvec,
              float* __restrict__ tail) {
  const int wave = threadIdx.x >> 5, lane = threadIdx.x & 31;
  const size_t row = (size_t)blockIdx.x * 8 + wave;  // over B*L
  const int b = (int)(row >> 12);                    // L = 4096
  const float* vr = V + row * HE + lane * 16;
  const float* cr = cvec + (size_t)b * HE + lane * 16;
  float dot = 0.f;
#pragma unroll
  for (int i = 0; i < 16; i += 4) {
    v4f v = *(const v4f*)(vr + i);
    v4f c = *(const v4f*)(cr + i);
    dot += v[0]*c[0] + v[1]*c[1] + v[2]*c[2] + v[3]*c[3];
  }
#pragma unroll
  for (int off = 16; off >= 1; off >>= 1) dot += __shfl_xor(dot, off, 32);
  if (lane == 0) tail[row] = 1.0f / ((float)HE + dot);
}

// ---------------------------------------------------------------------------
// Kernel 4: part = V @ KQ fused with epilogue
//   out = q + gamma * (vsum[m] + part[l,m]) * tailor[l]
// wave per 16x16 tile; 8 waves share one m-tile. The shared B panel (16 rows
// of KQ = contiguous 16KB) is staged ONCE per block into LDS via CDNA5 async
// global->LDS DMA (ASYNCcnt), with a padded row stride (1040B = 65*16B) so the
// 16 lanes' ds_load_b128 fragment reads are bank-conflict-free.
// grid (B, L/128, HE/16), block 256 = 8 waves (wave w -> l-tile w).
// ---------------------------------------------------------------------------
__global__ __launch_bounds__(256)
void part_k(const float* __restrict__ V, const bf16* __restrict__ KQ,
            const float* __restrict__ vsum, const float* __restrict__ tail,
            const float* __restrict__ gamma_p, float* __restrict__ out) {
  constexpr int KP = 520;                 // padded LDS row stride in bf16 (1040B)
  __shared__ bf16 ldsKQ[16 * KP];         // ~16.6 KB B panel

  const int b = blockIdx.x;
  const int wave = threadIdx.x >> 5, lane = threadIdx.x & 31;
  const int lr = lane & 15, hh = lane >> 4;
  const int l0 = blockIdx.y * 128 + wave * 16;
  const int m0 = blockIdx.z * 16;

  // ---- async-stage the 16x512 bf16 KQ panel (rows m0..m0+15) into LDS ----
  {
    const char* gpanel = (const char*)(KQ + ((size_t)b * HE + m0) * HE);
    const unsigned ldsbase = (unsigned)(size_t)(&ldsKQ[0]);
#pragma unroll
    for (int k = 0; k < 4; ++k) {
      const unsigned c    = (unsigned)threadIdx.x + k * 256;  // 16B chunk id
      const unsigned row  = c >> 6;                           // 0..15
      const unsigned col  = c & 63;                           // 16B chunk in row
      const unsigned loff = ldsbase + row * (KP * 2) + col * 16;
      const unsigned long long gaddr =
          (unsigned long long)(size_t)(gpanel + row * 1024 + col * 16);
      asm volatile("global_load_async_to_lds_b128 %0, %1, off"
                   :: "v"(loff), "v"(gaddr)
                   : "memory");
    }
    wait_asynccnt0();
    __syncthreads();
  }

  const float* Vrow = V + ((size_t)b * Ll + l0 + lr) * HE;  // A row (M = lr)
  const bf16*  Brow = &ldsKQ[lr * KP];                      // B col via symmetry

  v8f acc = {0.f,0.f,0.f,0.f,0.f,0.f,0.f,0.f};
#pragma unroll 4
  for (int ks = 0; ks < HE; ks += 32) {
    const int ka = ks + hh * 8;
    if (ks + 32 < HE) __builtin_prefetch(&Vrow[ka + 32], 0, 3);
    v4f a0 = *(const v4f*)&Vrow[ka];
    v4f a1 = *(const v4f*)&Vrow[ka + 4];
    v4f a2 = *(const v4f*)&Vrow[ka + 16];
    v4f a3 = *(const v4f*)&Vrow[ka + 20];
    v16bf a;
#pragma unroll
    for (int i = 0; i < 4; ++i) {
      a[i]      = (bf16)a0[i];
      a[4 + i]  = (bf16)a1[i];
      a[8 + i]  = (bf16)a2[i];
      a[12 + i] = (bf16)a3[i];
    }
    const int kb = ks + hh * 8;
    v8bf b0 = *(const v8bf*)&Brow[kb];
    v8bf b1 = *(const v8bf*)&Brow[kb + 16];
    v16bf bb = cat8(b0, b1);
    acc = __builtin_amdgcn_wmma_f32_16x16x32_bf16(false, a, false, bb,
                                                  (short)0, acc, false, false);
  }

  const float g  = gamma_p[0];
  const int   m  = m0 + lr;
  const float vs = vsum[b * HE + m];
#pragma unroll
  for (int r = 0; r < 8; ++r) {
    const int l = l0 + r + 8 * hh;
    const float tl = tail[(size_t)b * Ll + l];
    const size_t idx = ((size_t)b * Ll + l) * HE + m;
    out[idx] = V[idx] + g * ((vs + acc[r]) * tl);
  }
}

// ---------------------------------------------------------------------------
extern "C" void kernel_launch(void* const* d_in, const int* in_sizes, int n_in,
                              void* d_out, int out_size, void* d_ws, size_t ws_size,
                              hipStream_t stream) {
  const float* q       = (const float*)d_in[0];  // queries [B,L,H,E]
  const float* gamma_p = (const float*)d_in[4];  // gamma [1]
  float* out = (float*)d_out;

  // workspace layout (floats): psum 64K | psumsq 64K | ninv 8K | vsum 8K |
  //                            cvec 8K | tailor 64K | KQ bf16 16*512*512
  float* ws     = (float*)d_ws;
  float* psum   = ws;
  float* psumsq = psum   + (size_t)Bb * 8 * HE;
  float* ninv   = psumsq + (size_t)Bb * 8 * HE;
  float* vsum   = ninv   + (size_t)Bb * HE;
  float* cvec   = vsum   + (size_t)Bb * HE;
  float* tail   = cvec   + (size_t)Bb * HE;
  bf16*  KQ     = (bf16*)(tail + (size_t)Bb * Ll);

  stats_part_k <<<dim3(Bb, 8),              128, 0, stream>>>(q, psum, psumsq);
  stats_final_k<<<dim3(Bb),                 128, 0, stream>>>(psum, psumsq, ninv, vsum, cvec);
  gram_k       <<<dim3(Bb, 36),             128, 0, stream>>>(q, ninv, KQ);
  tailor_k     <<<dim3(Bb * Ll / 8),        256, 0, stream>>>(q, cvec, tail);
  part_k       <<<dim3(Bb, Ll/128, HE/16),  256, 0, stream>>>(q, KQ, vsum, tail, gamma_p, out);
}
